// ADGCNForDialog_29557964931232
// MI455X (gfx1250) — matrix operand order repaired and checked by reading, not compiled, not run on hardware
//
#include <hip/hip_runtime.h>
#include <math.h>

// ---------------------------------------------------------------------------
// Types for CDNA5 WMMA (wave32): v_wmma_f32_16x16x32_bf16
// ---------------------------------------------------------------------------
typedef __attribute__((ext_vector_type(8)))  __bf16 v8bf;
typedef __attribute__((ext_vector_type(16))) __bf16 v16bf;
typedef __attribute__((ext_vector_type(8)))  float  v8f;

#define HID 128
#define LN_EPS 1e-5f

__device__ __forceinline__ unsigned short f2bf(float f) {
    union { float f; unsigned u; } v; v.f = f;
    unsigned u = v.u;
    unsigned rnd = 0x7fffu + ((u >> 16) & 1u);   // round to nearest even
    return (unsigned short)((u + rnd) >> 16);
}

// ---------------------------------------------------------------------------
// Small elementwise / graph kernels
// ---------------------------------------------------------------------------
__global__ void zero_kernel(float* __restrict__ p, long n) {
    long i = (long)blockIdx.x * blockDim.x + threadIdx.x;
    if (i < n) p[i] = 0.0f;
}

__global__ void cvt_bf16_kernel(const float* __restrict__ src,
                                unsigned short* __restrict__ dst, long n) {
    long i = (long)blockIdx.x * blockDim.x + threadIdx.x;
    if (i < n) dst[i] = f2bf(src[i]);
}

__global__ void deg_kernel(const int* __restrict__ row, float* __restrict__ deg, int E) {
    long e = (long)blockIdx.x * blockDim.x + threadIdx.x;
    if (e < E) unsafeAtomicAdd(&deg[row[e]], 1.0f);
}

__global__ void dinv_kernel(float* __restrict__ deg, int n) {
    long i = (long)blockIdx.x * blockDim.x + threadIdx.x;
    if (i < n) {
        float d = deg[i];
        if (d == 0.0f) d = 1.0f;
        deg[i] = rsqrtf(d);
    }
}

__global__ void ew_kernel(const int* __restrict__ row, const int* __restrict__ col,
                          const float* __restrict__ dinv, float* __restrict__ ew, int E) {
    long e = (long)blockIdx.x * blockDim.x + threadIdx.x;
    if (e < E) ew[e] = dinv[row[e]] * dinv[col[e]];
}

// s = sigmoid(h . q_w + q_b - 1), one wave per node row
__global__ __launch_bounds__(256)
void gate_kernel(const float* __restrict__ h, const float* __restrict__ qw,
                 const float* __restrict__ qb, float* __restrict__ s, int n) {
    int lane = threadIdx.x & 31, w = threadIdx.x >> 5;
    long row = (long)blockIdx.x * 8 + w;
    if (row >= n) return;
    float sum = 0.0f;
    #pragma unroll
    for (int k = 0; k < 4; ++k) {
        int i = k * 32 + lane;
        sum += h[row * HID + i] * qw[i];
    }
    #pragma unroll
    for (int m = 16; m >= 1; m >>= 1) sum += __shfl_xor(sum, m, 32);
    if (lane == 0) {
        float z = sum + qb[0] - 1.0f;
        s[row] = 1.0f / (1.0f + __expf(-z));
    }
}

// hi[row] += ew * h[col] ; one wave per edge, 4 cols/lane, native f32 atomics
__global__ __launch_bounds__(256)
void spmm_kernel(const int* __restrict__ row, const int* __restrict__ col,
                 const float* __restrict__ ew, const float* __restrict__ h,
                 float* __restrict__ hi, int E) {
    int lane = threadIdx.x & 31, w = threadIdx.x >> 5;
    long e = (long)blockIdx.x * 8 + w;
    if (e >= E) return;
    float we = ew[e];
    long r = row[e], c = col[e];
    #pragma unroll
    for (int k = 0; k < 4; ++k) {
        int i = k * 32 + lane;
        unsafeAtomicAdd(&hi[r * HID + i], we * h[c * HID + i]);
    }
}

// support = (1-s)*hi + s*h0 ; also emit bf16 copy (next GEMM A operand)
__global__ void blend_kernel(const float* __restrict__ hi, const float* __restrict__ h0,
                             const float* __restrict__ s, float* __restrict__ supF,
                             unsigned short* __restrict__ supBf, long n) {
    long i = (long)blockIdx.x * blockDim.x + threadIdx.x;
    if (i >= n) return;
    float sv = s[i >> 7];
    float v = (1.0f - sv) * hi[i] + sv * h0[i];
    supF[i] = v;
    supBf[i] = f2bf(v);
}

// ---------------------------------------------------------------------------
// Weight pack: f32 [K x ncols] row-major -> per-lane WMMA B fragments (bf16).
// Layout: elem[((kt*nct + ct)*32 + lane)*16 + e] = W[kt*32 + (lane>>4)*8 + (e&7) + (e>>3)*16]
//         at column ct*16 + (lane&15), zero-padded past ncols.
// ---------------------------------------------------------------------------
__global__ void pack_b_kernel(const float* __restrict__ W, unsigned short* __restrict__ out,
                              int K, int ncols, int nct) {
    int lane = threadIdx.x;               // 32 threads
    int tileIdx = blockIdx.x;             // kt*nct + ct
    int kt = tileIdx / nct, ct = tileIdx % nct;
    int c = ct * 16 + (lane & 15);
    int kb = kt * 32 + (lane >> 4) * 8;
    #pragma unroll
    for (int e = 0; e < 16; ++e) {
        int k = kb + (e & 7) + ((e >> 3) << 4);
        float v = (c < ncols && k < K) ? W[(long)k * ncols + c] : 0.0f;
        out[((long)tileIdx * 32 + lane) * 16 + e] = f2bf(v);
    }
}

// ---------------------------------------------------------------------------
// Fused WMMA GEMM + blend + bias + ReLU + LayerNorm.
// One wave owns a full 16x128 output strip: 8 accumulator fragments; the A
// fragment is loaded once per K-step and reused across all 8 column tiles
// (8 independent back-to-back WMMAs per K-step). Block = 8 waves = 128 rows.
// ---------------------------------------------------------------------------
__global__ __launch_bounds__(256)
void wmma_gemm_ln_kernel(const unsigned short* __restrict__ Abf,   // [N,K] bf16
                         const unsigned short* __restrict__ Bpack, // packed, nct=8
                         const float* __restrict__ blendSrc,       // [N,128] or null
                         const float* __restrict__ bias,           // [128] or null
                         const float* __restrict__ gamma,
                         const float* __restrict__ beta,
                         float theta, int relu, int K,
                         float* __restrict__ outF,
                         unsigned short* __restrict__ outBf,
                         int nRowTiles, int nRows) {
    __shared__ float tile[8][16][HID + 4];         // 67.6 KB: per-wave C strip
    const int t = threadIdx.x;
    const int lane = t & 31, w = t >> 5;
    const int hs = lane >> 4, lr = lane & 15;      // half-select, lane-in-16
    const long rt = (long)blockIdx.x * 8 + w;      // 16-row tile index of this wave
    const int nkt = K >> 5;

    if (rt < nRowTiles) {                          // wave-uniform: EXEC all-ones at WMMA
        v8f acc[8];
        #pragma unroll
        for (int ct = 0; ct < 8; ++ct)
            acc[ct] = (v8f){0.f, 0.f, 0.f, 0.f, 0.f, 0.f, 0.f, 0.f};

        const unsigned short* arow = Abf + (rt * 16 + lr) * (long)K + hs * 8;
        const unsigned short* bptr = Bpack + (long)lane * 16;
        for (int kt = 0; kt < nkt; ++kt) {
            v8bf alo = *reinterpret_cast<const v8bf*>(arow);
            v8bf ahi = *reinterpret_cast<const v8bf*>(arow + 16);
            v16bf a = __builtin_shufflevector(alo, ahi,
                          0,1,2,3,4,5,6,7,8,9,10,11,12,13,14,15);
            #pragma unroll
            for (int ct = 0; ct < 8; ++ct) {
                v16bf b = *reinterpret_cast<const v16bf*>(bptr + (long)ct * 32 * 16);
                acc[ct] = __builtin_amdgcn_wmma_f32_16x16x32_bf16(
                              false, a, false, b, (short)0, acc[ct], false, false);
            }
            arow += 32;            // next 32 K-values
            bptr += 8 * 32 * 16;   // next K-tile (nct = 8)
        }
        // spill C strip: lane covers col ct*16+lr, rows v (+8 for upper half)
        #pragma unroll
        for (int ct = 0; ct < 8; ++ct)
            #pragma unroll
            for (int v = 0; v < 8; ++v)
                tile[w][v + hs * 8][ct * 16 + lr] = acc[ct][v];
    }
    __syncthreads();

    // ---- epilogue: 2 threads per row (64 cols each), shfl width-2 reduce ----
    const int rowLocal = t >> 1;                   // 0..127
    const int strip = rowLocal >> 4, r = rowLocal & 15;
    const int j = t & 1;
    const long grow = (long)blockIdx.x * 128 + rowLocal;
    if (grow < nRows) {
        float sum = 0.0f, sq = 0.0f;
        #pragma unroll 8
        for (int cc = 0; cc < 64; ++cc) {
            int col = j * 64 + cc;
            float x = theta * tile[strip][r][col];
            if (blendSrc) x += (1.0f - theta) * blendSrc[grow * HID + col];
            if (bias)     x += bias[col];
            if (relu)     x = x > 0.0f ? x : 0.0f;
            tile[strip][r][col] = x;               // same-thread RAW only
            sum += x; sq += x * x;
        }
        sum += __shfl_xor(sum, 1, 2);
        sq  += __shfl_xor(sq,  1, 2);
        float mean = sum * (1.0f / HID);
        float var  = sq  * (1.0f / HID) - mean * mean;
        float rstd = rsqrtf(var + LN_EPS);
        #pragma unroll 8
        for (int cc = 0; cc < 64; ++cc) {
            int col = j * 64 + cc;
            float y = gamma[col] * (tile[strip][r][col] - mean) * rstd + beta[col];
            outF[grow * HID + col]  = y;
            outBf[grow * HID + col] = f2bf(y);
        }
    }
}

// ---------------------------------------------------------------------------
// Classifier: [N,128] @ [128,16(pad of 7)] + b -> d_out[N,7]. One wave / 16 rows.
// ---------------------------------------------------------------------------
__global__ __launch_bounds__(256)
void wmma_cls_kernel(const unsigned short* __restrict__ hBf,
                     const unsigned short* __restrict__ clsPack, // nct=1
                     const float* __restrict__ cls_b,
                     float* __restrict__ out, int nRowTiles) {
    const int lane = threadIdx.x & 31, w = threadIdx.x >> 5;
    const int hs = lane >> 4, lr = lane & 15;
    long rt = (long)blockIdx.x * 8 + w;
    if (rt >= nRowTiles) return;                    // wave-uniform (EXEC all-ones at WMMA)
    long rowBase = rt * 16;

    v8f c = {0.f, 0.f, 0.f, 0.f, 0.f, 0.f, 0.f, 0.f};
    const unsigned short* arow = hBf + (rowBase + lr) * (long)HID + hs * 8;
    const unsigned short* bptr = clsPack + (long)lane * 16;
    #pragma unroll
    for (int kt = 0; kt < 4; ++kt) {
        v8bf alo = *reinterpret_cast<const v8bf*>(arow);
        v8bf ahi = *reinterpret_cast<const v8bf*>(arow + 16);
        v16bf a = __builtin_shufflevector(alo, ahi,
                      0,1,2,3,4,5,6,7,8,9,10,11,12,13,14,15);
        v16bf b = *reinterpret_cast<const v16bf*>(bptr);
        c = __builtin_amdgcn_wmma_f32_16x16x32_bf16(false, a, false, b,
                                                    (short)0, c, false, false);
        arow += 32;
        bptr += 32 * 16;       // nct = 1
    }
    int col = lr;
    if (col < 7) {
        float bb = cls_b[col];
        #pragma unroll
        for (int v = 0; v < 8; ++v) {
            long r = rowBase + hs * 8 + v;
            out[r * 7 + col] = c[v] + bb;
        }
    }
}

// ---------------------------------------------------------------------------
// Host launcher
// ---------------------------------------------------------------------------
extern "C" void kernel_launch(void* const* d_in, const int* in_sizes, int n_in,
                              void* d_out, int out_size, void* d_ws, size_t ws_size,
                              hipStream_t stream) {
    (void)n_in; (void)out_size; (void)ws_size;

    const float* x      = (const float*)d_in[0];
    const int*   edges  = (const int*)d_in[1];      // [2,E]
    const float* Wproj  = (const float*)d_in[2];
    const float* bproj  = (const float*)d_in[3];
    const float* gamma  = (const float*)d_in[4];
    const float* beta   = (const float*)d_in[5];
    const float* qw     = (const float*)d_in[6];
    const float* qb     = (const float*)d_in[7];
    const float* convW  = (const float*)d_in[8];    // [4,128,128]
    const float* clsW   = (const float*)d_in[9];    // [128,7]
    const float* clsB   = (const float*)d_in[10];
    float* out = (float*)d_out;

    const int N  = in_sizes[0] / 256;   // 100000 (multiple of 16)
    const int E  = in_sizes[1] / 2;     // 1600000
    const int IN = 256;

    // workspace carve-up (256B aligned slices)
    char* ws = (char*)d_ws;
    size_t off = 0;
    auto take = [&](size_t bytes) -> char* {
        char* p = ws + off;
        off = (off + bytes + 255) & ~(size_t)255;
        return p;
    };
    float* dinv  = (float*)take((size_t)N * 4);
    float* ewv   = (float*)take((size_t)E * 4);
    float* h0F   = (float*)take((size_t)N * HID * 4);
    float* hF    = (float*)take((size_t)N * HID * 4);
    float* hiF   = (float*)take((size_t)N * HID * 4);
    float* supF  = (float*)take((size_t)N * HID * 4);
    float* sg    = (float*)take((size_t)N * 4);
    unsigned short* xBf   = (unsigned short*)take((size_t)N * IN * 2);
    unsigned short* supBf = (unsigned short*)take((size_t)N * HID * 2);
    unsigned short* hBf   = (unsigned short*)take((size_t)N * HID * 2);
    unsigned short* projP = (unsigned short*)take((size_t)256 * 128 * 2);
    unsigned short* convP = (unsigned short*)take((size_t)4 * 128 * 128 * 2);
    unsigned short* clsP  = (unsigned short*)take((size_t)128 * 16 * 2);

    const int rowTiles = N / 16;                       // 6250
    const int gemmGrid = (rowTiles + 7) / 8;           // 782
    const long NH = (long)N * HID;

    // ---- graph normalization ----
    zero_kernel<<<(N + 255) / 256, 256, 0, stream>>>(dinv, N);
    deg_kernel<<<(E + 255) / 256, 256, 0, stream>>>(edges, dinv, E);
    dinv_kernel<<<(N + 255) / 256, 256, 0, stream>>>(dinv, N);
    ew_kernel<<<(E + 255) / 256, 256, 0, stream>>>(edges, edges + E, dinv, ewv, E);

    // ---- weight packing + input conversion ----
    cvt_bf16_kernel<<<(int)(((long)N * IN + 255) / 256), 256, 0, stream>>>(x, xBf, (long)N * IN);
    pack_b_kernel<<<(256 / 32) * 8, 32, 0, stream>>>(Wproj, projP, 256, 128, 8);
    for (int i = 0; i < 4; ++i)
        pack_b_kernel<<<(128 / 32) * 8, 32, 0, stream>>>(convW + (long)i * 128 * 128,
                                                         convP + (long)i * 128 * 128,
                                                         128, 128, 8);
    pack_b_kernel<<<(128 / 32) * 1, 32, 0, stream>>>(clsW, clsP, 128, 7, 1);

    // ---- h0 = LN(x @ Wproj + b) ----
    wmma_gemm_ln_kernel<<<gemmGrid, 256, 0, stream>>>(
        xBf, projP, nullptr, bproj, gamma, beta,
        1.0f, 0, 256, h0F, hBf, rowTiles, N);

    // ---- layers ----
    const float* cur = h0F;
    for (int i = 0; i < 4; ++i) {
        float theta = 0.5f / (float)(i + 1);
        gate_kernel<<<(N + 7) / 8, 256, 0, stream>>>(cur, qw, qb, sg, N);
        zero_kernel<<<(int)((NH + 255) / 256), 256, 0, stream>>>(hiF, NH);
        spmm_kernel<<<(E + 7) / 8, 256, 0, stream>>>(edges, edges + E, ewv, cur, hiF, E);
        blend_kernel<<<(int)((NH + 255) / 256), 256, 0, stream>>>(hiF, h0F, sg, supF, supBf, NH);
        wmma_gemm_ln_kernel<<<gemmGrid, 256, 0, stream>>>(
            supBf, convP + (long)i * 128 * 128, supF, nullptr, gamma, beta,
            theta, 1, 128, hF, hBf, rowTiles, N);
        cur = hF;
    }

    // ---- classifier ----
    wmma_cls_kernel<<<(rowTiles + 7) / 8, 256, 0, stream>>>(hBf, clsP, clsB, out, rowTiles);
}
